// WindowAttention_51659866636288
// MI455X (gfx1250) — matrix operand order repaired and checked by reading, not compiled, not run on hardware
//
#include <hip/hip_runtime.h>

// ---------------- types ----------------
typedef __bf16 bf16;
typedef __attribute__((ext_vector_type(16))) __bf16 v16bf;
typedef __attribute__((ext_vector_type(8)))  __bf16 v8bf;
typedef __attribute__((ext_vector_type(8)))  float  v8f;

#define DIM   384
#define TDIM  1152
#define HEADS 12
#define HD    32
#define NTOK  49
#define NP    64
#define SCALE 0.17677669529663689f   // 1/sqrt(32)

// LDS row strides (elements) — keep rows 16B-aligned and bank-spread
#define LDX  392   // sX, sO  (784 B rows)
#define LDQK 40    // sQ, sK  (80 B rows)
#define LDPV 72    // sP, sVt (144 B rows)

__device__ __forceinline__ v8f zero8() {
  v8f z = {0.f, 0.f, 0.f, 0.f, 0.f, 0.f, 0.f, 0.f};
  return z;
}

// ---- WMMA operand loaders (CDNA5 16x16x32 bf16 layouts) ----
// A (16x32, MxK): lane L holds row M=L&15; lanes 0-15 K=0..7,16..23;
// lanes 16-31 K=8..15,24..31.  Source is row-major [m][k] with stride ld.
__device__ __forceinline__ v16bf load_a(const bf16* base, int ld, int m0, int k0) {
  int lane = threadIdx.x & 31;
  const bf16* p = base + (size_t)(m0 + (lane & 15)) * ld + k0 + ((lane >> 4) << 3);
  v8bf lo = *(const v8bf*)p;
  v8bf hi = *(const v8bf*)(p + 16);
  v16bf a;
#pragma unroll
  for (int i = 0; i < 8; ++i) { a[i] = lo[i]; a[i + 8] = hi[i]; }
  return a;
}

// B (32x16, KxN): lane L holds column N=L&15; lanes 0-15 K=0..7,16..23,
// lanes 16-31 K=8..15,24..31.  Source is N-major (i.e. B^T row-major [n][k], stride ld)
// so each lane reads two runs of 8 contiguous bf16, 16 apart (two b128 loads).
__device__ __forceinline__ v16bf load_b(const bf16* bt, int ld, int n0, int k0) {
  int lane = threadIdx.x & 31;
  const bf16* p = bt + (size_t)(n0 + (lane & 15)) * ld + k0 + ((lane >> 4) << 3);
  v8bf lo = *(const v8bf*)p;
  v8bf hi = *(const v8bf*)(p + 16);
  v16bf b;
#pragma unroll
  for (int i = 0; i < 8; ++i) { b[i] = lo[i]; b[i + 8] = hi[i]; }
  return b;
}

__device__ __forceinline__ v8f wmma_bf16(v16bf a, v16bf b, v8f c) {
  return __builtin_amdgcn_wmma_f32_16x16x32_bf16(false, a, false, b, (short)0, c,
                                                 false, false);
}

// C/D (16x16 f32): VGPR i -> row m0+i (+8 for lanes 16-31), col n0+(lane&15).
__device__ __forceinline__ void store_c_bf16(bf16* dst, int ld, int m0, int n0, v8f c) {
  int lane = threadIdx.x & 31;
  int col  = n0 + (lane & 15);
  int rowb = m0 + ((lane >> 4) << 3);
#pragma unroll
  for (int i = 0; i < 8; ++i)
    dst[(size_t)(rowb + i) * ld + col] = (bf16)c[i];
}

// transposed store (for V^T staging): dst[n][m]
__device__ __forceinline__ void store_c_bf16_T(bf16* dst, int ld, int m0, int n0, v8f c) {
  int lane = threadIdx.x & 31;
  int row  = n0 + (lane & 15);
  int colb = m0 + ((lane >> 4) << 3);
#pragma unroll
  for (int i = 0; i < 8; ++i)
    dst[(size_t)row * ld + colb + i] = (bf16)c[i];
}

// ---------------- prep kernels (run once per launch, ~1.4 MB) ----------------
__global__ void prep_wqkv(const float* __restrict__ w, bf16* __restrict__ wt) {
  int idx = blockIdx.x * 256 + threadIdx.x;   // over TDIM*DIM, wt[oc][ic]
  if (idx >= TDIM * DIM) return;
  int oc = idx / DIM, ic = idx % DIM;
  wt[idx] = (bf16)w[(size_t)ic * TDIM + oc];
}

__global__ void prep_wproj(const float* __restrict__ w, bf16* __restrict__ wt) {
  int idx = blockIdx.x * 256 + threadIdx.x;   // over DIM*DIM, wt[oc][ic]
  if (idx >= DIM * DIM) return;
  int oc = idx / DIM, ic = idx % DIM;
  wt[idx] = (bf16)w[(size_t)ic * DIM + oc];
}

__global__ void prep_bias(const float* __restrict__ table, const int* __restrict__ relidx,
                          float* __restrict__ bp) {
  int idx = blockIdx.x * 256 + threadIdx.x;   // over HEADS*NP*NP
  if (idx >= HEADS * NP * NP) return;
  int h = idx / (NP * NP);
  int r = idx % (NP * NP);
  int i = r / NP, j = r % NP;
  float v = -1e30f;                           // mask pad columns pre-softmax
  if (i < NTOK && j < NTOK) v = table[relidx[i * NTOK + j] * HEADS + h];
  bp[idx] = v;
}

// ---------------- fused window-attention kernel ----------------
__global__ __launch_bounds__(256)
void winattn_kernel(const float* __restrict__ x,
                    const float* __restrict__ b_qkv,
                    const float* __restrict__ b_proj,
                    const bf16*  __restrict__ wqkv_t,   // [1152][384] bf16, N-major
                    const bf16*  __restrict__ wproj_t,  // [384][384]  bf16, N-major
                    const float* __restrict__ biasp,    // [12][64][64] f32, padded
                    float* __restrict__ out) {
  __shared__ bf16 sX [NP * LDX];          // x tile (bf16), rows >=49 zero
  __shared__ bf16 sO [NP * LDX];          // per-window attention output (all heads)
  __shared__ bf16 sQ [2 * NP * LDQK];     // per-head-pair Q (scaled)
  __shared__ bf16 sK [2 * NP * LDQK];     // per-head-pair K (token-major)
  __shared__ bf16 sVt[2 * HD * LDPV];     // per-head-pair V^T (d-major)
  __shared__ bf16 sP [2 * NP * LDPV];     // per-head-pair softmax probs

  const int b    = blockIdx.x;
  const int tid  = threadIdx.x;
  const int wave = tid >> 5;
  const int lane = tid & 31;

  // ---- Phase 0: load x window, fp32 -> bf16 into LDS (pad rows to zero) ----
  for (int idx = tid; idx < NP * (DIM / 4); idx += 256) {
    int row = idx / (DIM / 4);
    int c4  = (idx % (DIM / 4)) * 4;
    float4 v = make_float4(0.f, 0.f, 0.f, 0.f);
    if (row < NTOK)
      v = *(const float4*)(x + ((size_t)b * NTOK + row) * DIM + c4);
    bf16* d = sX + (size_t)row * LDX + c4;
    d[0] = (bf16)v.x; d[1] = (bf16)v.y; d[2] = (bf16)v.z; d[3] = (bf16)v.w;
  }
  __syncthreads();

  // ---- head-pair loop: 12 heads, 2 per iteration ----
  for (int hp = 0; hp < 6; ++hp) {
    // Sub-phase A: QKV projection for both heads (48 tiles over 8 waves)
    for (int j = 0; j < 6; ++j) {
      int t    = wave * 6 + j;      // 0..47
      int hh   = t / 24;            // head within pair
      int r    = t % 24;
      int mat  = r / 8;             // 0=q 1=k 2=v
      int mt   = (r % 8) / 2;
      int nt   = r % 2;
      int head = hp * 2 + hh;
      int chan0 = mat * DIM + head * HD + nt * 16;   // output-channel base
      const bf16* wt = wqkv_t + (size_t)chan0 * DIM;
      v8f acc = zero8();
#pragma unroll
      for (int ks = 0; ks < 12; ++ks) {
        v16bf a  = load_a(sX, LDX, mt * 16, ks * 32);
        v16bf bb = load_b(wt, DIM, 0, ks * 32);
        acc = wmma_bf16(a, bb, acc);
      }
      float bias = b_qkv[chan0 + (lane & 15)];
      if (mat == 0) {
#pragma unroll
        for (int i = 0; i < 8; ++i) acc[i] = (acc[i] + bias) * SCALE;
        store_c_bf16(sQ + hh * NP * LDQK, LDQK, mt * 16, nt * 16, acc);
      } else {
#pragma unroll
        for (int i = 0; i < 8; ++i) acc[i] += bias;
        if (mat == 1) store_c_bf16  (sK  + hh * NP * LDQK, LDQK, mt * 16, nt * 16, acc);
        else          store_c_bf16_T(sVt + hh * HD * LDPV, LDPV, mt * 16, nt * 16, acc);
      }
    }
    __syncthreads();

    // Sub-phase B: S = Q K^T (+bias), softmax -> sP.  wave -> (head, M-tile)
    {
      int hh = wave >> 2;
      int mt = wave & 3;
      int head = hp * 2 + hh;
      const bf16* q  = sQ + hh * NP * LDQK;
      const bf16* kk = sK + hh * NP * LDQK;
      v8f s[4];
#pragma unroll
      for (int nt = 0; nt < 4; ++nt)
        s[nt] = wmma_bf16(load_a(q, LDQK, mt * 16, 0),
                          load_b(kk, LDQK, nt * 16, 0), zero8());
      int colb = lane & 15;
      int rowb = mt * 16 + ((lane >> 4) << 3);
      const float* bp = biasp + ((size_t)head * NP + rowb) * NP + colb;
#pragma unroll
      for (int nt = 0; nt < 4; ++nt)
#pragma unroll
        for (int i = 0; i < 8; ++i)
          s[nt][i] += bp[i * NP + nt * 16];

      bf16* p = sP + hh * NP * LDPV;
#pragma unroll
      for (int i = 0; i < 8; ++i) {
        float mx = fmaxf(fmaxf(s[0][i], s[1][i]), fmaxf(s[2][i], s[3][i]));
        mx = fmaxf(mx, __shfl_xor(mx, 1, 32));
        mx = fmaxf(mx, __shfl_xor(mx, 2, 32));
        mx = fmaxf(mx, __shfl_xor(mx, 4, 32));
        mx = fmaxf(mx, __shfl_xor(mx, 8, 32));   // reduce within 16-lane half
        float e0 = __expf(s[0][i] - mx);
        float e1 = __expf(s[1][i] - mx);
        float e2 = __expf(s[2][i] - mx);
        float e3 = __expf(s[3][i] - mx);
        float sum = e0 + e1 + e2 + e3;
        sum += __shfl_xor(sum, 1, 32);
        sum += __shfl_xor(sum, 2, 32);
        sum += __shfl_xor(sum, 4, 32);
        sum += __shfl_xor(sum, 8, 32);
        float rinv = 1.0f / sum;
        size_t ro = (size_t)(rowb + i) * LDPV + colb;
        p[ro]      = (bf16)(e0 * rinv);
        p[ro + 16] = (bf16)(e1 * rinv);
        p[ro + 32] = (bf16)(e2 * rinv);
        p[ro + 48] = (bf16)(e3 * rinv);
      }
    }
    __syncthreads();

    // Sub-phase C: O = P V  (wave owns same (head, M-tile) slab)
    {
      int hh = wave >> 2;
      int mt = wave & 3;
      int head = hp * 2 + hh;
      const bf16* p  = sP  + hh * NP * LDPV;
      const bf16* vt = sVt + hh * HD * LDPV;
#pragma unroll
      for (int nt = 0; nt < 2; ++nt) {
        v8f acc = zero8();
#pragma unroll
        for (int ks = 0; ks < 2; ++ks)
          acc = wmma_bf16(load_a(p, LDPV, mt * 16, ks * 32),
                          load_b(vt, LDPV, nt * 16, ks * 32), acc);
        store_c_bf16(sO, LDX, mt * 16, head * HD + nt * 16, acc);
      }
    }
    __syncthreads();
  }

  // ---- Final projection: out = O @ W_proj + b_proj (96 tiles over 8 waves) ----
  for (int j = 0; j < 12; ++j) {
    int t  = wave * 12 + j;  // 0..95
    int mt = t / 24;
    int nt = t % 24;
    v8f acc = zero8();
#pragma unroll
    for (int ks = 0; ks < 12; ++ks)
      acc = wmma_bf16(load_a(sO, LDX, mt * 16, ks * 32),
                      load_b(wproj_t, DIM, nt * 16, ks * 32), acc);
    int col  = nt * 16 + (lane & 15);
    float bias = b_proj[col];
    int rowb = mt * 16 + ((lane >> 4) << 3);
#pragma unroll
    for (int i = 0; i < 8; ++i) {
      int row = rowb + i;
      if (row < NTOK)
        out[((size_t)b * NTOK + row) * DIM + col] = acc[i] + bias;
    }
  }
}

// ---------------- launch ----------------
extern "C" void kernel_launch(void* const* d_in, const int* in_sizes, int n_in,
                              void* d_out, int out_size, void* d_ws, size_t ws_size,
                              hipStream_t stream) {
  const float* x      = (const float*)d_in[0];
  const float* W_qkv  = (const float*)d_in[1];
  const float* b_qkv  = (const float*)d_in[2];
  const float* W_proj = (const float*)d_in[3];
  const float* b_proj = (const float*)d_in[4];
  const float* table  = (const float*)d_in[5];
  const int*   relidx = (const int*)d_in[6];
  float* out = (float*)d_out;

  const int B = in_sizes[0] / (NTOK * DIM);

  // workspace layout (all 256B aligned)
  char* ws = (char*)d_ws;
  bf16*  wqkv_t  = (bf16*)ws;                                  // 1152*384*2 = 884736
  bf16*  wproj_t = (bf16*)(ws + 884736);                       // 384*384*2  = 294912
  float* biasp   = (float*)(ws + 884736 + 294912);             // 12*64*64*4 = 196608

  prep_wqkv <<<(TDIM * DIM + 255) / 256, 256, 0, stream>>>(W_qkv, wqkv_t);
  prep_wproj<<<(DIM * DIM  + 255) / 256, 256, 0, stream>>>(W_proj, wproj_t);
  prep_bias <<<(HEADS * NP * NP + 255) / 256, 256, 0, stream>>>(table, relidx, biasp);

  winattn_kernel<<<B, 256, 0, stream>>>(x, b_qkv, b_proj, wqkv_t, wproj_t, biasp, out);
}
